// encoder_chain_11218454577218
// MI455X (gfx1250) — matrix operand order repaired and checked by reading, not compiled, not run on hardware
//
#include <hip/hip_runtime.h>
#include <hip/hip_bf16.h>
#include <climits>

typedef __attribute__((ext_vector_type(16))) _Float16 v16h;
typedef __attribute__((ext_vector_type(8)))  float    v8f;

#define TPB 256

// ---------------------------------------------------------------------------
// Fill kernels (graph-capture-safe init; no hipMemset)
// ---------------------------------------------------------------------------
__global__ void fill_f32(float* __restrict__ p, float v, long n) {
  long i = (long)blockIdx.x * blockDim.x + threadIdx.x;
  if (i < n) p[i] = v;
}
__global__ void fill_i32(int* __restrict__ p, int v, long n) {
  long i = (long)blockIdx.x * blockDim.x + threadIdx.x;
  if (i < n) p[i] = v;
}

// ---------------------------------------------------------------------------
// WMMA GEMM: Z[M,Nout] = X[M,K] @ W[Nout,K]^T   (f32 in, f16 WMMA, f32 acc/out)
//
// K is a template parameter (64 or 128) so the K-loop fully unrolls and the
// v_wmma_f32_16x16x32_f16 ops issue back-to-back. Each wave computes a
// 16(M) x 32(N) tile: two 16x16 WMMAs per K-step share one A fragment
// (halves A loads/converts; enables XDL co-issue).
//
// Fragment layouts (wave32):
//  A 16x32 f16: lane l (m=l&15, hs=l>>4) holds K = hs*16+[0..15]
//  B 32x16 f16: lane l holds column n=l&15, K = hs*16+[0..15]
//  C/D f32:     lane l vgpr j -> row = hs*8+j, col = l&15
//
// M%16==0 (N=50000), Nout%32==0 (128/64/256), K%32==0: no edge handling.
// All f32 sub-rows are 16B-aligned -> float4 (b128) operand loads.
// tileM guard is wave-uniform, so EXEC is all-1s at every WMMA.
// ---------------------------------------------------------------------------
__device__ __forceinline__ void cvt4(v16h& v, int o, float4 f) {
  v[o]     = (_Float16)f.x; v[o + 1] = (_Float16)f.y;
  v[o + 2] = (_Float16)f.z; v[o + 3] = (_Float16)f.w;
}

template <int K>
__global__ void gemm_xwT_wmma(const float* __restrict__ X, const float* __restrict__ W,
                              float* __restrict__ Z, int M, int Nout) {
  const int lane = threadIdx.x & 31;
  const int wave = threadIdx.x >> 5;
  const int tileM = blockIdx.x * (blockDim.x >> 5) + wave;
  const int tileN = blockIdx.y;                 // 32-wide N tile
  if (tileM * 16 >= M) return;                  // wave-uniform
  const int hs = lane >> 4;                     // 0 or 1
  const int mr = lane & 15;
  const float* arow  = X + (long)(tileM * 16 + mr) * K;
  const float* brow0 = W + (long)(tileN * 32 + mr) * K;
  const float* brow1 = W + (long)(tileN * 32 + 16 + mr) * K;
  v8f c0 = {}, c1 = {};
#pragma unroll
  for (int kb = 0; kb < K; kb += 32) {
    v16h a, b0, b1;
    {
      const float4* ap = (const float4*)(arow + kb + hs * 16);
      cvt4(a, 0, ap[0]); cvt4(a, 4, ap[1]);          // K = hs*16 + [0..7]
      const float4* ap2 = (const float4*)(arow + kb + hs * 16 + 8);
      cvt4(a, 8, ap2[0]); cvt4(a, 12, ap2[1]);       // K = hs*16 + 8 + [0..7]
    }
    {
      const float4* bp = (const float4*)(brow0 + kb + hs * 16);
#pragma unroll
      for (int q = 0; q < 4; ++q) cvt4(b0, 4 * q, bp[q]);
    }
    {
      const float4* bp = (const float4*)(brow1 + kb + hs * 16);
#pragma unroll
      for (int q = 0; q < 4; ++q) cvt4(b1, 4 * q, bp[q]);
    }
    c0 = __builtin_amdgcn_wmma_f32_16x16x32_f16(false, a, false, b0, (short)0, c0,
                                                false, false);
    c1 = __builtin_amdgcn_wmma_f32_16x16x32_f16(false, a, false, b1, (short)0, c1,
                                                false, false);
  }
#pragma unroll
  for (int j = 0; j < 8; ++j) {
    long rr = (long)(tileM * 16 + hs * 8 + j) * Nout + tileN * 32 + mr;
    Z[rr]      = c0[j];
    Z[rr + 16] = c1[j];
  }
}

// ---------------------------------------------------------------------------
// Attention logits: el[n,h] = sum_d z[n,h,d]*al[h,d];  er likewise.
// z layout: [N, H*D] contiguous (matches reshape(n,H,D)).
// ---------------------------------------------------------------------------
__global__ void el_er_kernel(const float* __restrict__ z, const float* __restrict__ al,
                             const float* __restrict__ ar, float* __restrict__ el,
                             float* __restrict__ er, int N, int H, int D) {
  int idx = blockIdx.x * blockDim.x + threadIdx.x;   // n*H + h
  if (idx >= N * H) return;
  int h = idx % H;
  const float* zp = z + (long)idx * D;
  const float* alp = al + h * D;
  const float* arp = ar + h * D;
  float sl = 0.f, sr = 0.f;
  for (int d = 0; d < D; ++d) { float v = zp[d]; sl += v * alp[d]; sr += v * arp[d]; }
  el[idx] = sl; er[idx] = sr;
}

// monotone float <-> int encoding for atomicMax on signed int
__device__ __forceinline__ int f2ord(float f) {
  int b = __float_as_int(f);
  return b >= 0 ? b : (b ^ 0x7fffffff);
}
__device__ __forceinline__ float ord2f(int k) {
  int b = k >= 0 ? k : (k ^ 0x7fffffff);
  return __int_as_float(b);
}

// pass1: e = leaky_relu(el[src]+er[dst], 0.2); segment-max into emax[dst]
__global__ void edge_pass1(const float* __restrict__ el, const float* __restrict__ er,
                           const int* __restrict__ src, const int* __restrict__ dst,
                           float* __restrict__ ebuf, int* __restrict__ emax,
                           int E, int H) {
  long idx = (long)blockIdx.x * blockDim.x + threadIdx.x;   // e*H + h
  if (idx >= (long)E * H) return;
  int e = (int)(idx / H), h = (int)(idx % H);
  int s = src[e], d = dst[e];
  float v = el[s * H + h] + er[d * H + h];
  v = v > 0.f ? v : 0.2f * v;
  ebuf[idx] = v;
  atomicMax(&emax[d * H + h], f2ord(v));
}

// pass2: ee = exp(e - emax[dst]); denominator via atomicAdd
__global__ void edge_pass2(float* __restrict__ ebuf, float* __restrict__ den,
                           const int* __restrict__ emax, const int* __restrict__ dst,
                           int E, int H) {
  long idx = (long)blockIdx.x * blockDim.x + threadIdx.x;
  if (idx >= (long)E * H) return;
  int e = (int)(idx / H), h = (int)(idx % H);
  int d = dst[e];
  float mx = ord2f(emax[d * H + h]);       // every edge's dst has >=1 edge -> valid
  float ee = expf(ebuf[idx] - mx);
  ebuf[idx] = ee;
  atomicAdd(&den[d * H + h], ee);
}

// pass3: rst[dst,h,:] += (ee/den[dst,h]) * z[src,h,:]   (thread per (edge,d), loop H)
__global__ void edge_pass3(const float* __restrict__ ebuf, const float* __restrict__ den,
                           const float* __restrict__ z, const int* __restrict__ src,
                           const int* __restrict__ dst, float* __restrict__ rst,
                           int E, int H, int D) {
  long idx = (long)blockIdx.x * blockDim.x + threadIdx.x;   // e*D + d
  if (idx >= (long)E * D) return;
  int e = (int)(idx / D), d = (int)(idx % D);
  int s = src[e], t = dst[e];
  for (int h = 0; h < H; ++h) {
    float alpha = ebuf[(long)e * H + h] / den[t * H + h];
    atomicAdd(&rst[((long)t * H + h) * D + d],
              alpha * z[((long)s * H + h) * D + d]);
  }
}

// HeteroGraphConv mean: out = 0.5*(rstA + bA + chainGAT(zB) + bB)
// chain etype has exactly one in-edge per node>0 -> softmax==1 -> shift(zB).
__global__ void combine_kernel(const float* __restrict__ rstA, const float* __restrict__ bA,
                               const float* __restrict__ zB, const float* __restrict__ bB,
                               float* __restrict__ out, int N, int HD) {
  long idx = (long)blockIdx.x * blockDim.x + threadIdx.x;   // n*HD + c
  if (idx >= (long)N * HD) return;
  int c = (int)(idx % HD);
  long n = idx / HD;
  float vb = (n > 0) ? zB[idx - HD] : 0.f;
  out[idx] = 0.5f * (rstA[idx] + bA[c] + vb + bB[c]);
}

// chain_pass: m[n,h,:] = in[n-1,h,:] - in[n,h,:] (0 for n=0), L2-normalized over D
__global__ void chain_pass_kernel(const float* __restrict__ in, float* __restrict__ out,
                                  int N, int H, int D) {
  long idx = (long)blockIdx.x * blockDim.x + threadIdx.x;   // n*H + h
  if (idx >= (long)N * H) return;
  int h = (int)(idx % H);
  long n = idx / H;
  long base = ((long)n * H + h) * D;
  if (n == 0) { for (int d = 0; d < D; ++d) out[base + d] = 0.f; return; }
  long pbase = base - (long)H * D;
  float ss = 0.f;
  for (int d = 0; d < D; ++d) { float m = in[pbase + d] - in[base + d]; ss += m * m; }
  float inv = 1.f / (sqrtf(ss) + 1e-7f);
  for (int d = 0; d < D; ++d) out[base + d] = (in[pbase + d] - in[base + d]) * inv;
}

// ---------------------------------------------------------------------------
// Chunked cumsum over nodes, 256 channels (threadIdx.x = channel, coalesced).
// ---------------------------------------------------------------------------
__global__ void scan_partial(const float* __restrict__ ds, float* __restrict__ partial,
                             int N, int L) {
  int c = threadIdx.x;
  long start = (long)blockIdx.x * L;
  long stop = start + L; if (stop > N) stop = N;
  float s = 0.f;
  for (long i = start; i < stop; ++i) s += ds[i * 256 + c];
  partial[blockIdx.x * 256 + c] = s;
}
__global__ void scan_offsets(float* __restrict__ partial, int CH) {
  int c = threadIdx.x;
  float run = 0.f;
  for (int b = 0; b < CH; ++b) {
    float t = partial[b * 256 + c];
    partial[b * 256 + c] = run;
    run += t;
  }
}
__global__ void scan_final(const float* __restrict__ ds, const float* __restrict__ partial,
                           float* __restrict__ out, int N, int L) {
  int c = threadIdx.x;
  long start = (long)blockIdx.x * L;
  long stop = start + L; if (stop > N) stop = N;
  float acc = partial[blockIdx.x * 256 + c];
  for (long i = start; i < stop; ++i) {
    acc += ds[i * 256 + c];
    out[i * 256 + c] = acc;
  }
}

// ---------------------------------------------------------------------------
// Host-side helpers
// ---------------------------------------------------------------------------
static inline int gblk(long n) { return (int)((n + TPB - 1) / TPB); }

static void launch_gemm(const float* X, const float* W, float* Z,
                        int M, int K, int Nout, hipStream_t stream) {
  int mtiles = (M + 15) / 16;
  dim3 grid((mtiles + 3) / 4, Nout / 32);
  if (K == 64) {
    gemm_xwT_wmma<64><<<grid, 128, 0, stream>>>(X, W, Z, M, Nout);
  } else {
    gemm_xwT_wmma<128><<<grid, 128, 0, stream>>>(X, W, Z, M, Nout);
  }
}

static void launch_gat_edges(const float* z, const float* al, const float* ar,
                             const int* src, const int* dst,
                             float* el, float* er, int* emaxi, float* den,
                             float* ebuf, float* rst,
                             int N, int E, int H, int D, hipStream_t stream) {
  long NH = (long)N * H;
  el_er_kernel<<<gblk(NH), TPB, 0, stream>>>(z, al, ar, el, er, N, H, D);
  fill_i32<<<gblk(NH), TPB, 0, stream>>>(emaxi, INT_MIN, NH);
  fill_f32<<<gblk(NH), TPB, 0, stream>>>(den, 0.f, NH);
  long rn = (long)N * H * D;
  fill_f32<<<gblk(rn), TPB, 0, stream>>>(rst, 0.f, rn);
  long eh = (long)E * H;
  edge_pass1<<<gblk(eh), TPB, 0, stream>>>(el, er, src, dst, ebuf, emaxi, E, H);
  edge_pass2<<<gblk(eh), TPB, 0, stream>>>(ebuf, den, emaxi, dst, E, H);
  long ed = (long)E * D;
  edge_pass3<<<gblk(ed), TPB, 0, stream>>>(ebuf, den, z, src, dst, rst, E, H, D);
}

extern "C" void kernel_launch(void* const* d_in, const int* in_sizes, int n_in,
                              void* d_out, int out_size, void* d_ws, size_t ws_size,
                              hipStream_t stream) {
  (void)n_in; (void)out_size; (void)ws_size;
  const float* x   = (const float*)d_in[0];
  const int* src0  = (const int*)d_in[1];
  const int* dst0  = (const int*)d_in[2];
  // src1/dst1 (d_in[3],d_in[4]) are the chain i->i+1: handled analytically.
  const float* W1a = (const float*)d_in[5];
  const float* al1a= (const float*)d_in[6];
  const float* ar1a= (const float*)d_in[7];
  const float* b1a = (const float*)d_in[8];
  const float* W1b = (const float*)d_in[9];
  const float* b1b = (const float*)d_in[12];
  const float* W2a = (const float*)d_in[13];
  const float* al2a= (const float*)d_in[14];
  const float* ar2a= (const float*)d_in[15];
  const float* b2a = (const float*)d_in[16];
  const float* W2b = (const float*)d_in[17];
  const float* b2b = (const float*)d_in[20];
  const float* Wma = (const float*)d_in[21];
  const float* alma= (const float*)d_in[22];
  const float* arma= (const float*)d_in[23];
  const float* bma = (const float*)d_in[24];
  const float* Wmb = (const float*)d_in[25];
  const float* bmb = (const float*)d_in[28];
  // d_in[29] = r: dead code in reference.

  const int N = in_sizes[0] / 64;   // 50000
  const int E = in_sizes[1];        // 400000

  // Workspace carve-out (floats): ~215 MB total
  float* ws    = (float*)d_ws;
  float* zA    = ws;                     // N*256
  float* zB    = zA    + (long)N * 256;  // N*256
  float* rst   = zB    + (long)N * 256;  // N*256
  float* feat  = rst   + (long)N * 256;  // N*128  (h after layer1)
  float* feat2 = feat  + (long)N * 128;  // N*64   (h2 after layer2)
  float* conh  = feat2 + (long)N * 64;   // N*64
  float* el    = conh  + (long)N * 64;   // N*4
  float* er    = el    + (long)N * 4;    // N*4
  float* den   = er    + (long)N * 4;    // N*4
  int*   emaxi = (int*)(den + (long)N * 4);          // N*4 ints
  float* ebuf  = (float*)(emaxi + (long)N * 4);      // E*4
  float* partial = ebuf + (long)E * 4;               // 200*256

  // ---- Layer 1: in=64 -> hid=128, H=1 ----
  launch_gemm(x, W1a, zA, N, 64, 128, stream);
  launch_gemm(x, W1b, zB, N, 64, 128, stream);
  launch_gat_edges(zA, al1a, ar1a, src0, dst0, el, er, emaxi, den, ebuf, rst,
                   N, E, 1, 128, stream);
  combine_kernel<<<gblk((long)N * 128), TPB, 0, stream>>>(rst, b1a, zB, b1b, feat, N, 128);

  // ---- Layer 2: hid=128 -> out=64, H=1 ----
  launch_gemm(feat, W2a, zA, N, 128, 64, stream);
  launch_gemm(feat, W2b, zB, N, 128, 64, stream);
  launch_gat_edges(zA, al2a, ar2a, src0, dst0, el, er, emaxi, den, ebuf, rst,
                   N, E, 1, 64, stream);
  combine_kernel<<<gblk((long)N * 64), TPB, 0, stream>>>(rst, b2a, zB, b2b, feat2, N, 64);

  // ---- chain_pass + cumsum over singleton head axis (= identity) -> conh ----
  chain_pass_kernel<<<gblk((long)N), TPB, 0, stream>>>(feat2, conh, N, 1, 64);

  // ---- MH layer: out=64 -> H=4, D=64 (Nout=256) ----
  launch_gemm(conh, Wma, zA, N, 64, 256, stream);
  launch_gemm(conh, Wmb, zB, N, 64, 256, stream);
  launch_gat_edges(zA, alma, arma, src0, dst0, el, er, emaxi, den, ebuf, rst,
                   N, E, 4, 64, stream);
  float* hm = zA;   // zA free after edge_pass3 (stream-ordered)
  combine_kernel<<<gblk((long)N * 256), TPB, 0, stream>>>(rst, bma, zB, bmb, hm, N, 256);

  // ---- per-head chain_pass -> ds (reuse zB), then cumsum over nodes -> d_out ----
  float* ds = zB;
  chain_pass_kernel<<<gblk((long)N * 4), TPB, 0, stream>>>(hm, ds, N, 4, 64);

  const int CH = 200;
  const int L  = (N + CH - 1) / CH;   // 250
  scan_partial<<<CH, 256, 0, stream>>>(ds, partial, N, L);
  scan_offsets<<<1, 256, 0, stream>>>(partial, CH);
  scan_final<<<CH, 256, 0, stream>>>(ds, partial, (float*)d_out, N, L);
}
